// TemporalCrossAttention_6399501271559
// MI455X (gfx1250) — compile-verified, matrix operand order checked by reading, not run on hardware
//
#include <hip/hip_runtime.h>

typedef __bf16 bf16;
typedef __attribute__((ext_vector_type(16))) __bf16 v16bf;
typedef __attribute__((ext_vector_type(8)))  __bf16 v8bf;
typedef __attribute__((ext_vector_type(8)))  float   v8f;

#define BB 128
#define TT 243
#define DD 1024
#define NTOK 77
#define LL 512
#define HH 16
#define MTOT (BB*TT)      // 31104 = 243*128
#define MKV  (BB*NTOK)    // 9856  = 77*128

__device__ inline v16bf cat8(v8bf lo, v8bf hi) {
    return __builtin_shufflevector(lo, hi, 0,1,2,3,4,5,6,7,8,9,10,11,12,13,14,15);
}
__device__ inline v8f v8f_zero() {
    v8f z;
#pragma unroll
    for (int i = 0; i < 8; i++) z[i] = 0.0f;
    return z;
}

// ---------------- transpose + f32->bf16 weight prep: dst[n*K+k] = src[k*N+n]
__global__ __launch_bounds__(256) void transpose_bf16_kernel(
    const float* __restrict__ src, bf16* __restrict__ dst, int K, int N) {
    __shared__ float tile[32][33];
    int bx = blockIdx.x, by = blockIdx.y;          // bx over N/32, by over K/32
    int tx = threadIdx.x & 31, ty = threadIdx.x >> 5;
#pragma unroll
    for (int i = 0; i < 4; i++) {
        int k = by * 32 + ty + i * 8;
        tile[ty + i * 8][tx] = src[(size_t)k * N + bx * 32 + tx];
    }
    __syncthreads();
#pragma unroll
    for (int i = 0; i < 4; i++) {
        int n = bx * 32 + ty + i * 8;
        dst[(size_t)n * K + by * 32 + tx] = (bf16)tile[tx][ty + i * 8];
    }
}

// ---------------- block reduce (sum, sumsq)
__device__ inline void block_reduce2(float& s, float& s2) {
    __shared__ float sh[16];
#pragma unroll
    for (int o = 16; o > 0; o >>= 1) { s += __shfl_xor(s, o); s2 += __shfl_xor(s2, o); }
    int w = threadIdx.x >> 5, l = threadIdx.x & 31;
    if (l == 0) { sh[w] = s; sh[w + 8] = s2; }
    __syncthreads();
    if (threadIdx.x < 32) {
        float a = (l < 8) ? sh[l] : 0.0f;
        float c = (l < 8) ? sh[l + 8] : 0.0f;
#pragma unroll
        for (int o = 4; o > 0; o >>= 1) { a += __shfl_xor(a, o); c += __shfl_xor(c, o); }
        if (l == 0) { sh[0] = a; sh[1] = c; }
    }
    __syncthreads();
    s = sh[0]; s2 = sh[1];
}

// ---------------- LayerNorm -> bf16 (one block per token)
__global__ __launch_bounds__(256) void ln_bf16_kernel(
    const float* __restrict__ x, const float* __restrict__ g,
    const float* __restrict__ b, bf16* __restrict__ out, int C) {
    const float* row = x + (size_t)blockIdx.x * C;
    float s = 0.0f, s2 = 0.0f;
    for (int i = threadIdx.x; i < C; i += 256) { float v = row[i]; s += v; s2 += v * v; }
    block_reduce2(s, s2);
    float mean = s / C;
    float var = s2 / C - mean * mean;
    float rstd = rsqrtf(var + 1e-5f);
    bf16* orow = out + (size_t)blockIdx.x * C;
    for (int i = threadIdx.x; i < C; i += 256)
        orow[i] = (bf16)((row[i] - mean) * rstd * g[i] + b[i]);
}

// ---------------- silu(emb[:,0,:]) -> bf16 (128*1024 elements)
__global__ __launch_bounds__(256) void silu_emb_kernel(
    const float* __restrict__ emb, bf16* __restrict__ se) {
    int i = blockIdx.x * 256 + threadIdx.x;       // [0, 131072)
    int b = i >> 10, d = i & 1023;
    float x = emb[(size_t)b * TT * DD + d];
    se[i] = (bf16)(x / (1.0f + __expf(-x)));
}

// ---------------- AdaLN: a = silu(LN2(y)*(1+scale)+shift) -> bf16
__global__ __launch_bounds__(256) void adaln_kernel(
    const float* __restrict__ y, const float* __restrict__ g,
    const float* __restrict__ bvec, const float* __restrict__ eout,
    bf16* __restrict__ a) {
    int id = blockIdx.x;
    int b = id / TT;
    const float* row = y + (size_t)id * DD;
    float s = 0.0f, s2 = 0.0f;
    for (int i = threadIdx.x; i < DD; i += 256) { float v = row[i]; s += v; s2 += v * v; }
    block_reduce2(s, s2);
    float mean = s / DD;
    float rstd = rsqrtf(s2 / DD - mean * mean + 1e-5f);
    bf16* arow = a + (size_t)id * DD;
    const float* sc = eout + (size_t)b * 2 * DD;
    for (int i = threadIdx.x; i < DD; i += 256) {
        float val = (row[i] - mean) * rstd * g[i] + bvec[i];
        val = val * (1.0f + sc[i]) + sc[DD + i];
        arow[i] = (bf16)(val / (1.0f + __expf(-val)));
    }
}

// ---------------- bf16 WMMA GEMM: C = A(MxK) * W^T(NxK) + bias
// MODE 0: bf16 out; MODE 1: f32 out + residual; MODE 2: f32 out
// Register-pipelined: next tile's global loads overlap current tile's WMMAs.
template <int MODE>
__global__ __launch_bounds__(256) void gemm_bf16_kernel(
    const bf16* __restrict__ A, const bf16* __restrict__ Wt,
    const float* __restrict__ bias, const float* __restrict__ res,
    float* __restrict__ outf, bf16* __restrict__ outb,
    int M, int K, int Nc) {
    __shared__ __attribute__((aligned(16))) bf16 As[128 * 64];
    __shared__ __attribute__((aligned(16))) bf16 Bs[128 * 64];
    const int m0 = blockIdx.x * 128;
    const int n0 = blockIdx.y * 128;
    const int tid = threadIdx.x;
    const int lane = tid & 31;
    const int w = tid >> 5;
    const int wr = w >> 1, wc = w & 1;
    const int l16 = lane & 15, lh = lane >> 4;

    v8f acc[2][4];
#pragma unroll
    for (int i = 0; i < 2; i++)
#pragma unroll
        for (int j = 0; j < 4; j++) acc[i][j] = v8f_zero();

    // prologue: fetch tile 0 into registers
    v8bf pa[4], pb[4];
#pragma unroll
    for (int g = 0; g < 4; g++) {
        int gi = tid + g * 256;                    // 1024 granules of 16B
        int row = gi >> 3, off = (gi & 7) * 8;
        pa[g] = *(const v8bf*)&A[(size_t)(m0 + row) * K + off];
        pb[g] = *(const v8bf*)&Wt[(size_t)(n0 + row) * K + off];
    }

    for (int k0 = 0; k0 < K; k0 += 64) {
        // dump prefetched tile into LDS
#pragma unroll
        for (int g = 0; g < 4; g++) {
            int gi = tid + g * 256;
            int row = gi >> 3, off = (gi & 7) * 8;
            *(v8bf*)&As[row * 64 + off] = pa[g];
            *(v8bf*)&Bs[row * 64 + off] = pb[g];
        }
        __syncthreads();
        // issue next tile's global loads (overlap with WMMAs below)
        if (k0 + 64 < K) {
#pragma unroll
            for (int g = 0; g < 4; g++) {
                int gi = tid + g * 256;
                int row = gi >> 3, off = (gi & 7) * 8;
                pa[g] = *(const v8bf*)&A[(size_t)(m0 + row) * K + k0 + 64 + off];
                pb[g] = *(const v8bf*)&Wt[(size_t)(n0 + row) * K + k0 + 64 + off];
            }
        }
#pragma unroll
        for (int ks = 0; ks < 2; ks++) {
            const int kb = lh * 8 + ks * 32;
            v16bf af[2], bfm[4];
#pragma unroll
            for (int ms = 0; ms < 2; ms++) {
                int row = wr * 32 + ms * 16 + l16;
                af[ms] = cat8(*(const v8bf*)&As[row * 64 + kb],
                              *(const v8bf*)&As[row * 64 + kb + 16]);
            }
#pragma unroll
            for (int ns = 0; ns < 4; ns++) {
                int col = wc * 64 + ns * 16 + l16;
                bfm[ns] = cat8(*(const v8bf*)&Bs[col * 64 + kb],
                               *(const v8bf*)&Bs[col * 64 + kb + 16]);
            }
#pragma unroll
            for (int ms = 0; ms < 2; ms++)
#pragma unroll
                for (int ns = 0; ns < 4; ns++)
                    acc[ms][ns] = __builtin_amdgcn_wmma_f32_16x16x32_bf16(
                        false, af[ms], false, bfm[ns], (short)0, acc[ms][ns],
                        false, false);
        }
        __syncthreads();
    }

#pragma unroll
    for (int ns = 0; ns < 4; ns++) {
        int n = n0 + wc * 64 + ns * 16 + l16;
        float bb = bias[n];
#pragma unroll
        for (int ms = 0; ms < 2; ms++) {
#pragma unroll
            for (int r = 0; r < 8; r++) {
                int m = m0 + wr * 32 + ms * 16 + r + lh * 8;
                size_t idx = (size_t)m * Nc + n;
                float v = acc[ms][ns][r] + bb;
                if (MODE == 0)       outb[idx] = (bf16)v;
                else if (MODE == 1)  outf[idx] = res[idx] + v;
                else                 outf[idx] = v;
            }
        }
    }
}

// ---------------- attention: one block per (b,h); y (B,T,H,dh) f32
__global__ __launch_bounds__(256) void attn_kernel(
    const bf16* __restrict__ q, const bf16* __restrict__ k,
    const bf16* __restrict__ v, float* __restrict__ y) {
    __shared__ __attribute__((aligned(16))) bf16 kbuf[80 * 64];
    __shared__ __attribute__((aligned(16))) bf16 vT[64 * 96];
    __shared__ __attribute__((aligned(16))) bf16 wbuf[8][16 * 96];
    const int bh = blockIdx.x;
    const int b = bh >> 4, h = bh & 15;
    const int tid = threadIdx.x, lane = tid & 31, w = tid >> 5;
    const int l16 = lane & 15, lh = lane >> 4;

    // stage k (80 tokens x 64, pad rows zeroed)
    for (int gi = tid; gi < 80 * 8; gi += 256) {
        int n = gi >> 3, off = (gi & 7) * 8;
        v8bf val = {};
        if (n < NTOK)
            val = *(const v8bf*)&k[((size_t)(b * NTOK + n) * DD) + h * 64 + off];
        *(v8bf*)&kbuf[n * 64 + off] = val;
    }
    // stage v transposed: vT[d][tok], tokens padded to 96 (zero)
    for (int gi = tid; gi < 64 * 96; gi += 256) {
        int d = gi / 96, n = gi % 96;
        bf16 val = (bf16)0.0f;
        if (n < NTOK) val = v[((size_t)(b * NTOK + n) * DD) + h * 64 + d];
        vT[d * 96 + n] = val;
    }
    // zero wbuf pad columns 80..95
    for (int gi = tid; gi < 8 * 16 * 16; gi += 256) {
        int wv = gi >> 8, row = (gi >> 4) & 15, c = gi & 15;
        wbuf[wv][row * 96 + 80 + c] = (bf16)0.0f;
    }
    __syncthreads();

    for (int it = 0; it < 2; it++) {
        const int t0 = (w + it * 8) * 16;
        // ---- QK^T  (M=16 rows, N=80 tokens, K=64)
        v8f s[5];
#pragma unroll
        for (int c = 0; c < 5; c++) s[c] = v8f_zero();
#pragma unroll
        for (int ks = 0; ks < 2; ks++) {
            const int kb = lh * 8 + ks * 32;
            int tq = t0 + l16; if (tq > TT - 1) tq = TT - 1;
            const bf16* qp = q + (size_t)(b * TT + tq) * DD + h * 64 + kb;
            v16bf af = cat8(*(const v8bf*)qp, *(const v8bf*)(qp + 16));
#pragma unroll
            for (int c = 0; c < 5; c++) {
                int col = c * 16 + l16;
                v16bf bfm = cat8(*(const v8bf*)&kbuf[col * 64 + kb],
                                 *(const v8bf*)&kbuf[col * 64 + kb + 16]);
                s[c] = __builtin_amdgcn_wmma_f32_16x16x32_bf16(
                    false, af, false, bfm, (short)0, s[c], false, false);
            }
        }
        // ---- softmax over tokens (cols across 16 lanes x 5 chunks)
#pragma unroll
        for (int r = 0; r < 8; r++) {
            float vals[5]; float m = -3.0e38f;
#pragma unroll
            for (int c = 0; c < 5; c++) {
                int col = c * 16 + l16;
                float sv = (col < NTOK) ? s[c][r] * 0.125f : -3.0e38f;
                vals[c] = sv; m = sv > m ? sv : m;
            }
#pragma unroll
            for (int o = 1; o < 16; o <<= 1) {
                float t = __shfl_xor(m, o); m = t > m ? t : m;
            }
            float sum = 0.0f;
#pragma unroll
            for (int c = 0; c < 5; c++) {
                float p = __expf(vals[c] - m); vals[c] = p; sum += p;
            }
#pragma unroll
            for (int o = 1; o < 16; o <<= 1) sum += __shfl_xor(sum, o);
            float inv = 1.0f / sum;
            int row = r + lh * 8;
#pragma unroll
            for (int c = 0; c < 5; c++)
                wbuf[w][row * 96 + c * 16 + l16] = (bf16)(vals[c] * inv);
        }
        __syncthreads();
        // ---- A*V  (M=16, N=64, K=96 padded)
        v8f ya[4];
#pragma unroll
        for (int nb = 0; nb < 4; nb++) ya[nb] = v8f_zero();
#pragma unroll
        for (int ks = 0; ks < 3; ks++) {
            const int kb = lh * 8 + ks * 32;
            v16bf af = cat8(*(const v8bf*)&wbuf[w][l16 * 96 + kb],
                            *(const v8bf*)&wbuf[w][l16 * 96 + kb + 16]);
#pragma unroll
            for (int nb = 0; nb < 4; nb++) {
                int d = nb * 16 + l16;
                v16bf bfm = cat8(*(const v8bf*)&vT[d * 96 + kb],
                                 *(const v8bf*)&vT[d * 96 + kb + 16]);
                ya[nb] = __builtin_amdgcn_wmma_f32_16x16x32_bf16(
                    false, af, false, bfm, (short)0, ya[nb], false, false);
            }
        }
#pragma unroll
        for (int nb = 0; nb < 4; nb++)
#pragma unroll
            for (int r = 0; r < 8; r++) {
                int t = t0 + r + lh * 8;
                if (t < TT)
                    y[(size_t)(b * TT + t) * DD + h * 64 + nb * 16 + l16] =
                        ya[nb][r];
            }
        __syncthreads();
    }
}

extern "C" void kernel_launch(void* const* d_in, const int* in_sizes, int n_in,
                              void* d_out, int out_size, void* d_ws, size_t ws_size,
                              hipStream_t stream) {
    const float* x    = (const float*)d_in[0];
    const float* xf   = (const float*)d_in[1];
    const float* emb  = (const float*)d_in[2];
    const float* ln_g = (const float*)d_in[3];
    const float* ln_b = (const float*)d_in[4];
    const float* tln_g= (const float*)d_in[5];
    const float* tln_b= (const float*)d_in[6];
    const float* Wq   = (const float*)d_in[7];
    const float* bq   = (const float*)d_in[8];
    const float* Wk   = (const float*)d_in[9];
    const float* bk   = (const float*)d_in[10];
    const float* Wv   = (const float*)d_in[11];
    const float* bv   = (const float*)d_in[12];
    const float* We   = (const float*)d_in[13];
    const float* be   = (const float*)d_in[14];
    const float* ln2_g= (const float*)d_in[15];
    const float* ln2_b= (const float*)d_in[16];
    const float* Wo   = (const float*)d_in[17];
    const float* bo   = (const float*)d_in[18];
    float* out = (float*)d_out;

    char* ws = (char*)d_ws;
    size_t off = 0;
    auto alloc = [&](size_t bytes) { char* p = ws + off; off = (off + bytes + 255) & ~(size_t)255; return p; };
    bf16* lnx  = (bf16*)alloc((size_t)MTOT * DD * 2);
    bf16* xfn  = (bf16*)alloc((size_t)MKV * LL * 2);
    bf16* qb   = (bf16*)alloc((size_t)MTOT * DD * 2);
    bf16* kb_  = (bf16*)alloc((size_t)MKV * DD * 2);
    bf16* vb   = (bf16*)alloc((size_t)MKV * DD * 2);
    float* yb  = (float*)alloc((size_t)MTOT * DD * 4);
    bf16* ab   = (bf16*)alloc((size_t)MTOT * DD * 2);
    bf16* seb  = (bf16*)alloc((size_t)BB * DD * 2);
    float* eout= (float*)alloc((size_t)BB * 2 * DD * 4);
    bf16* WqT  = (bf16*)alloc((size_t)DD * DD * 2);
    bf16* WkT  = (bf16*)alloc((size_t)DD * LL * 2);
    bf16* WvT  = (bf16*)alloc((size_t)DD * LL * 2);
    bf16* WeT  = (bf16*)alloc((size_t)2 * DD * DD * 2);
    bf16* WoT  = (bf16*)alloc((size_t)DD * DD * 2);

    dim3 blk(256);
    // weight prep (transpose + bf16)
    transpose_bf16_kernel<<<dim3(DD/32, DD/32), blk, 0, stream>>>(Wq, WqT, DD, DD);
    transpose_bf16_kernel<<<dim3(DD/32, LL/32), blk, 0, stream>>>(Wk, WkT, LL, DD);
    transpose_bf16_kernel<<<dim3(DD/32, LL/32), blk, 0, stream>>>(Wv, WvT, LL, DD);
    transpose_bf16_kernel<<<dim3(2*DD/32, DD/32), blk, 0, stream>>>(We, WeT, DD, 2*DD);
    transpose_bf16_kernel<<<dim3(DD/32, DD/32), blk, 0, stream>>>(Wo, WoT, DD, DD);
    // layernorms + silu(emb0)
    ln_bf16_kernel<<<MTOT, blk, 0, stream>>>(x, ln_g, ln_b, lnx, DD);
    ln_bf16_kernel<<<MKV, blk, 0, stream>>>(xf, tln_g, tln_b, xfn, LL);
    silu_emb_kernel<<<(BB * DD) / 256, blk, 0, stream>>>(emb, seb);
    // projections
    gemm_bf16_kernel<0><<<dim3(MTOT/128, DD/128), blk, 0, stream>>>(
        lnx, WqT, bq, nullptr, nullptr, qb, MTOT, DD, DD);
    gemm_bf16_kernel<0><<<dim3(MKV/128, DD/128), blk, 0, stream>>>(
        xfn, WkT, bk, nullptr, nullptr, kb_, MKV, LL, DD);
    gemm_bf16_kernel<0><<<dim3(MKV/128, DD/128), blk, 0, stream>>>(
        xfn, WvT, bv, nullptr, nullptr, vb, MKV, LL, DD);
    gemm_bf16_kernel<2><<<dim3(1, 2*DD/128), blk, 0, stream>>>(
        seb, WeT, be, nullptr, eout, nullptr, BB, DD, 2*DD);
    // attention
    attn_kernel<<<BB * HH, blk, 0, stream>>>(qb, kb_, vb, yb);
    // AdaLN + silu
    adaln_kernel<<<MTOT, blk, 0, stream>>>(yb, ln2_g, ln2_b, eout, ab);
    // out-proj + residual -> d_out
    gemm_bf16_kernel<1><<<dim3(MTOT/128, DD/128), blk, 0, stream>>>(
        ab, WoT, bo, x, out, nullptr, MTOT, DD, DD);
}
